// TransposedAttention_33921651704441
// MI455X (gfx1250) — matrix operand order repaired and checked by reading, not compiled, not run on hardware
//
#include <hip/hip_runtime.h>
#include <hip/hip_bf16.h>

// Transposed (channel) attention for x:(4,256,128,128) f32.
//   qkv = W(768,256) @ x(b,256,16384) + bias
//   attn = softmax_j( q_i . k_j / 16 )            (256x256 per batch)
//   out  = attn @ v + x
// GEMMs on v_wmma_f32_16x16x32_bf16 (wave32).
//   K1: sync staging (f32->bf16 convert) + global_prefetch
//   K2: GLOBAL_LOAD_ASYNC_TO_LDS_B128 + ASYNCcnt, double buffered
//   K4: TDM tensor_load_to_lds + TENSORcnt, double buffered (async fallback)

#define C_IN  256
#define OC    768
#define NPIX  16384
#define NBATCH 4

typedef __attribute__((ext_vector_type(16))) __bf16 v16bf;
typedef __attribute__((ext_vector_type(8)))  float  v8f;
typedef __attribute__((ext_vector_type(4)))  int    v4i;
typedef __attribute__((ext_vector_type(4)))  unsigned int v4u;
typedef __attribute__((ext_vector_type(8)))  int    v8i;

union FragBF {
    v16bf v;
    uint4 u[2];
};

// ---- async global->LDS staging (ASYNCcnt) ---------------------------------
#if defined(__AMDGCN__) && __has_builtin(__builtin_amdgcn_global_load_async_to_lds_b128) && __has_builtin(__builtin_amdgcn_s_wait_asynccnt)
#define USE_ASYNC_LDS 1
#else
#define USE_ASYNC_LDS 0
#if defined(__AMDGCN__)
#warning "gfx1250 async-to-LDS builtins unavailable: sync fallback"
#endif
#endif

#if USE_ASYNC_LDS
typedef __attribute__((address_space(1))) v4i as1_v4i;   // global
typedef __attribute__((address_space(3))) v4i as3_v4i;   // LDS
#endif

__device__ __forceinline__ void async_copy_b128(const __bf16* g, __bf16* l) {
#if USE_ASYNC_LDS
    __builtin_amdgcn_global_load_async_to_lds_b128((as1_v4i*)g, (as3_v4i*)l, 0, 0);
#else
    *(uint4*)l = *(const uint4*)g;
#endif
}

__device__ __forceinline__ void wait_async() {
#if USE_ASYNC_LDS
    __builtin_amdgcn_s_wait_asynccnt(0);
#endif
}

// ---- TDM tensor DMA staging (TENSORcnt) -----------------------------------
#if defined(__AMDGCN__) && __has_builtin(__builtin_amdgcn_tensor_load_to_lds) && __has_builtin(__builtin_amdgcn_s_wait_tensorcnt)
#define USE_TDM 1
#else
#define USE_TDM 0
#if defined(__AMDGCN__)
#warning "gfx1250 TDM builtins unavailable: async fallback for kernel 4"
#endif
#endif

__device__ __forceinline__ unsigned lds_offset_bytes(const void* p) {
    return (unsigned)(size_t)(__attribute__((address_space(3))) const void*)p;
}

// 2D bf16 tile load via Tensor Data Mover (D# per ISA ch8).
// tile_d0 elems per row (contiguous), tile_d1 rows, row stride in elems.
// LDS destination padded: pad_amount+1 DWORDs inserted every (1<<pad_interval)
// 8-byte units (programmed to match the padded LDS row strides).
__device__ __forceinline__ void tdm_load_2d(const void* gsrc, unsigned lds_off,
                                            unsigned tile_d0, unsigned tile_d1,
                                            unsigned stride0_elems,
                                            int pad_interval, int pad_amount) {
#if USE_TDM
    const unsigned long long ga = (unsigned long long)(size_t)gsrc;
    v4u g0;
    g0.x = 1u;                                               // count=1, user mode
    g0.y = lds_off;                                          // lds_addr (bytes)
    g0.z = (unsigned)(ga & 0xFFFFFFFFu);                     // global_addr[31:0]
    g0.w = (unsigned)((ga >> 32) & 0x1FFFFFFu) | (2u << 30); // addr[56:32] | type=2

    v8i g1;
    g1[0] = (1 << 16)                    // data_size: 2 bytes (bf16)
          | (1 << 20)                    // pad_enable
          | (pad_interval << 22)
          | (pad_amount << 25);
    const unsigned td0 = tile_d0;        // tensor_dim0 == tile_dim0 (no OOB)
    const unsigned td1 = tile_d1;        // tensor_dim1 == tile_dim1
    g1[1] = (int)((td0 & 0xFFFFu) << 16);                  // tensor_dim0[15:0]
    g1[2] = (int)((td0 >> 16) | ((td1 & 0xFFFFu) << 16));  // dim0 hi | dim1 lo
    g1[3] = (int)((td1 >> 16) | (tile_d0 << 16));          // dim1 hi | tile_dim0
    g1[4] = (int)(tile_d1 & 0xFFFFu);                      // tile_dim1 (tile_dim2=0)
    g1[5] = (int)stride0_elems;                            // tensor_dim0_stride lo
    g1[6] = 0;                                             // stride hi / dim1_stride
    g1[7] = 0;

    const v4i z4 = {0, 0, 0, 0};                           // 2D: groups 2/3 unused
    const v8i z8 = {0, 0, 0, 0, 0, 0, 0, 0};
    __builtin_amdgcn_tensor_load_to_lds(g0, g1, z4, z4, z8, 0);
#endif
}

__device__ __forceinline__ void wait_tensor() {
#if USE_TDM
    __builtin_amdgcn_s_wait_tensorcnt(0);
#endif
}

// ---- fragment loaders (ISA 7.12.2 layouts) ---------------------------------
__device__ __forceinline__ v16bf load_a_frag(const __bf16* row_base, int lane,
                                             int row_stride, int m_off) {
    const int mrow = lane & 15;
    const int sub  = (lane < 16) ? 0 : 8;
    const __bf16* r = row_base + (size_t)(m_off + mrow) * row_stride;
    FragBF f;
    f.u[0] = *(const uint4*)(r + sub);
    f.u[1] = *(const uint4*)(r + 16 + sub);
    return f.v;
}

__device__ __forceinline__ v16bf load_b_frag(const __bf16* row_base, int lane,
                                             int row_stride, int n_off) {
    const int sub = (lane < 16) ? 0 : 8;
    const __bf16* r = row_base + n_off + (lane & 15);
    FragBF f;
#pragma unroll
    for (int i = 0; i < 8; ++i) {
        f.v[i]     = r[(size_t)(sub + i) * row_stride];
        f.v[8 + i] = r[(size_t)(16 + sub + i) * row_stride];
    }
    return f.v;
}

#define WMMA_BF16(A, B, Cacc) \
    __builtin_amdgcn_wmma_f32_16x16x32_bf16(false, (A), false, (B), (short)0, (Cacc), false, false)

// ============================================================================
// Kernel 1: qkv = W @ x + b   (M=768, K=256, N=16384 per batch), bf16 out.
// ============================================================================
__global__ __launch_bounds__(256) void qkv_gemm(const float* __restrict__ x,
                                                const float* __restrict__ w,
                                                const float* __restrict__ bias,
                                                __bf16* __restrict__ qkv) {
    const int n0 = blockIdx.x * 64;
    const int o0 = blockIdx.y * 128;
    const int b  = blockIdx.z;

    __shared__ __bf16 As[128][40];
    __shared__ __bf16 Bs[32][80];

    const int tid  = threadIdx.x;
    const int lane = tid & 31;
    const int wave = tid >> 5;
    const int wm   = (wave & 3) * 32;
    const int wn   = (wave >> 2) * 32;

    const float* xb = x + (size_t)b * C_IN * NPIX;

    const int arow = tid >> 1;
    const int acb  = (tid & 1) * 16;
    const int brow = tid >> 3;
    const int bcb  = (tid & 7) * 8;

    v8f acc[2][2] = {};

    for (int k0 = 0; k0 < C_IN; k0 += 32) {
        {
            const float* src = w + (size_t)(o0 + arow) * C_IN + k0 + acb;
#pragma unroll
            for (int i = 0; i < 16; i += 4) {
                float4 f = *(const float4*)(src + i);
                As[arow][acb + i + 0] = (__bf16)f.x;
                As[arow][acb + i + 1] = (__bf16)f.y;
                As[arow][acb + i + 2] = (__bf16)f.z;
                As[arow][acb + i + 3] = (__bf16)f.w;
            }
        }
        {
            const float* src = xb + (size_t)(k0 + brow) * NPIX + n0 + bcb;
            float4 f0 = *(const float4*)(src);
            float4 f1 = *(const float4*)(src + 4);
            Bs[brow][bcb + 0] = (__bf16)f0.x; Bs[brow][bcb + 1] = (__bf16)f0.y;
            Bs[brow][bcb + 2] = (__bf16)f0.z; Bs[brow][bcb + 3] = (__bf16)f0.w;
            Bs[brow][bcb + 4] = (__bf16)f1.x; Bs[brow][bcb + 5] = (__bf16)f1.y;
            Bs[brow][bcb + 6] = (__bf16)f1.z; Bs[brow][bcb + 7] = (__bf16)f1.w;
        }
        if (k0 + 32 < C_IN) {   // WGP-scope prefetch of next K tile
            __builtin_prefetch(w + (size_t)(o0 + arow) * C_IN + k0 + 32 + acb, 0, 3);
            __builtin_prefetch(xb + (size_t)(k0 + 32 + brow) * NPIX + n0 + bcb, 0, 3);
        }
        __syncthreads();

        v16bf a0 = load_a_frag(&As[0][0], lane, 40, wm);
        v16bf a1 = load_a_frag(&As[0][0], lane, 40, wm + 16);
        v16bf b0 = load_b_frag(&Bs[0][0], lane, 80, wn);
        v16bf b1 = load_b_frag(&Bs[0][0], lane, 80, wn + 16);

        acc[0][0] = WMMA_BF16(a0, b0, acc[0][0]);
        acc[0][1] = WMMA_BF16(a0, b1, acc[0][1]);
        acc[1][0] = WMMA_BF16(a1, b0, acc[1][0]);
        acc[1][1] = WMMA_BF16(a1, b1, acc[1][1]);
        __syncthreads();
    }

#pragma unroll
    for (int tm = 0; tm < 2; ++tm) {
#pragma unroll
        for (int tn = 0; tn < 2; ++tn) {
#pragma unroll
            for (int r = 0; r < 8; ++r) {
                const int oo = o0 + wm + tm * 16 + ((lane < 16) ? r : r + 8);
                const int nn = n0 + wn + tn * 16 + (lane & 15);
                const float v = acc[tm][tn][r] + bias[oo];
                qkv[((size_t)b * OC + oo) * NPIX + nn] = (__bf16)v;
            }
        }
    }
}

// ============================================================================
// Kernel 2: S += scale * q @ k^T  (256x256 per batch, K=16384, 8-way K split)
// Q tile async to LDS; K tile transposed through VGPRs. Double buffered.
// ============================================================================
__global__ __launch_bounds__(256) void qk_scores(const __bf16* __restrict__ qkv,
                                                 float* __restrict__ attn) {
    const int j0 = blockIdx.x * 64;
    const int i0 = blockIdx.y * 64;
    const int b  = blockIdx.z >> 3;
    const int kc = blockIdx.z & 7;

    const __bf16* q  = qkv + (size_t)b * OC * NPIX;
    const __bf16* kp = qkv + ((size_t)b * OC + C_IN) * NPIX;

    __shared__ __bf16 Qs[2][64][40];
    __shared__ __bf16 Kt[2][32][80];

    const int tid  = threadIdx.x;
    const int lane = tid & 31;
    const int wave = tid >> 5;
    const int wm   = (wave & 3) * 16;
    const int wn   = (wave >> 2) * 32;

    const int row = tid >> 2;
    const int cb  = (tid & 3) * 8;

    v8f acc[2] = {};

    const int kbeg = kc * (NPIX / 8);
    const int kend = kbeg + (NPIX / 8);

    auto stage = [&](int k0, int buf) {
        async_copy_b128(q + (size_t)(i0 + row) * NPIX + k0 + cb, &Qs[buf][row][cb]);
        union { uint4 u; __bf16 h[8]; } t;
        t.u = *(const uint4*)(kp + (size_t)(j0 + row) * NPIX + k0 + cb);
#pragma unroll
        for (int i = 0; i < 8; ++i) Kt[buf][cb + i][row] = t.h[i];
    };

    int cur = 0;
    stage(kbeg, 0);
    for (int k0 = kbeg; k0 < kend; k0 += 32) {
        wait_async();
        __syncthreads();
        if (k0 + 32 < kend) stage(k0 + 32, cur ^ 1);

        v16bf a0 = load_a_frag(&Qs[cur][0][0], lane, 40, wm);
        v16bf b0 = load_b_frag(&Kt[cur][0][0], lane, 80, wn);
        v16bf b1 = load_b_frag(&Kt[cur][0][0], lane, 80, wn + 16);

        acc[0] = WMMA_BF16(a0, b0, acc[0]);
        acc[1] = WMMA_BF16(a0, b1, acc[1]);
        cur ^= 1;
    }

    const float scale = 0.0625f;  // 1/sqrt(256)
#pragma unroll
    for (int tn = 0; tn < 2; ++tn) {
#pragma unroll
        for (int r = 0; r < 8; ++r) {
            const int i = i0 + wm + ((lane < 16) ? r : r + 8);
            const int j = j0 + wn + tn * 16 + (lane & 15);
            atomicAdd(&attn[((size_t)b * C_IN + i) * C_IN + j], acc[tn][r] * scale);
        }
    }
}

// ============================================================================
// Kernel 3: row softmax over attn (1024 rows of 256), f32 -> bf16
// ============================================================================
__global__ __launch_bounds__(256) void softmax_rows(const float* __restrict__ attn,
                                                    __bf16* __restrict__ attn_bf) {
    const int row = blockIdx.x;
    const int tid = threadIdx.x;
    __shared__ float red[256];

    const float v = attn[(size_t)row * C_IN + tid];

    red[tid] = v;
    __syncthreads();
    for (int s = 128; s > 0; s >>= 1) {
        if (tid < s) red[tid] = fmaxf(red[tid], red[tid + s]);
        __syncthreads();
    }
    const float m = red[0];
    __syncthreads();

    const float e = __expf(v - m);
    red[tid] = e;
    __syncthreads();
    for (int s = 128; s > 0; s >>= 1) {
        if (tid < s) red[tid] += red[tid + s];
        __syncthreads();
    }
    const float inv = 1.0f / red[0];

    attn_bf[(size_t)row * C_IN + tid] = (__bf16)(e * inv);
}

// ============================================================================
// Kernel 4: out = attn @ v + x   (M=256, K=256, N=16384 per batch)
// Tiles staged by the Tensor Data Mover (wave 0 issues, TENSORcnt + barrier),
// double buffered. Falls back to ASYNCcnt staging if TDM builtin is absent.
// ============================================================================
__global__ __launch_bounds__(256) void av_out(const __bf16* __restrict__ attn_bf,
                                              const __bf16* __restrict__ qkv,
                                              const float* __restrict__ x,
                                              float* __restrict__ out) {
    const int n0 = blockIdx.x * 64;
    const int i0 = blockIdx.y * 128;
    const int b  = blockIdx.z;

    const __bf16* A = attn_bf + (size_t)b * C_IN * C_IN;            // [i][j]
    const __bf16* V = qkv + ((size_t)b * OC + 2 * C_IN) * NPIX;     // [j][n]

    __shared__ __bf16 As[2][128][40];   // rows: 64B data + 16B pad
    __shared__ __bf16 Bs[2][32][80];    // rows: 128B data + 32B pad

    const int tid  = threadIdx.x;
    const int lane = tid & 31;
    const int wave = tid >> 5;
    const int wm   = (wave & 3) * 32;
    const int wn   = (wave >> 2) * 32;

    const int arow = tid >> 1;
    const int acb  = (tid & 1) * 16;
    const int brow = tid >> 3;
    const int bcb  = (tid & 7) * 8;

    v8f acc[2][2] = {};

    auto stage = [&](int k0, int buf) {
#if USE_TDM
        if (wave == 0) {
            // attn tile: 32 elems/row x 128 rows, row stride 256 elems
            // LDS pad: every 16 DWORDs (64B) insert 4 DWORDs (16B): intv=3, amt=3
            tdm_load_2d(A + (size_t)i0 * C_IN + k0,
                        lds_offset_bytes(&As[buf][0][0]),
                        32, 128, C_IN, 3, 3);
            // v tile: 64 elems/row x 32 rows, row stride 16384 elems
            // LDS pad: every 32 DWORDs (128B) insert 8 DWORDs (32B): intv=4, amt=7
            tdm_load_2d(V + (size_t)k0 * NPIX + n0,
                        lds_offset_bytes(&Bs[buf][0][0]),
                        64, 32, NPIX, 4, 7);
        }
#else
        const __bf16* srcA = A + (size_t)(i0 + arow) * C_IN + k0 + acb;
        async_copy_b128(srcA,     &As[buf][arow][acb]);
        async_copy_b128(srcA + 8, &As[buf][arow][acb + 8]);
        async_copy_b128(V + (size_t)(k0 + brow) * NPIX + n0 + bcb, &Bs[buf][brow][bcb]);
#endif
    };

    int cur = 0;
    stage(0, 0);
    for (int k0 = 0; k0 < C_IN; k0 += 32) {
#if USE_TDM
        if (wave == 0) wait_tensor();
#else
        wait_async();
#endif
        __syncthreads();
        if (k0 + 32 < C_IN) stage(k0 + 32, cur ^ 1);

        v16bf a0 = load_a_frag(&As[cur][0][0], lane, 40, wm);
        v16bf a1 = load_a_frag(&As[cur][0][0], lane, 40, wm + 16);
        v16bf b0 = load_b_frag(&Bs[cur][0][0], lane, 80, wn);
        v16bf b1 = load_b_frag(&Bs[cur][0][0], lane, 80, wn + 16);

        acc[0][0] = WMMA_BF16(a0, b0, acc[0][0]);
        acc[0][1] = WMMA_BF16(a0, b1, acc[0][1]);
        acc[1][0] = WMMA_BF16(a1, b0, acc[1][0]);
        acc[1][1] = WMMA_BF16(a1, b1, acc[1][1]);
        cur ^= 1;
    }

#pragma unroll
    for (int tm = 0; tm < 2; ++tm) {
#pragma unroll
        for (int tn = 0; tn < 2; ++tn) {
#pragma unroll
            for (int r = 0; r < 8; ++r) {
                const int ii = i0 + wm + tm * 16 + ((lane < 16) ? r : r + 8);
                const int nn = n0 + wn + tn * 16 + (lane & 15);
                const size_t idx = ((size_t)b * C_IN + ii) * NPIX + nn;
                out[idx] = acc[tm][tn][r] + x[idx];
            }
        }
    }
}

// ============================================================================
extern "C" void kernel_launch(void* const* d_in, const int* in_sizes, int n_in,
                              void* d_out, int out_size, void* d_ws, size_t ws_size,
                              hipStream_t stream) {
    const float* x    = (const float*)d_in[0];
    const float* w    = (const float*)d_in[1];
    const float* bias = (const float*)d_in[2];
    float* out        = (float*)d_out;

    __bf16* qkv        = (__bf16*)d_ws;                                   // 96 MiB
    const size_t qkv_bytes  = (size_t)NBATCH * OC * NPIX * sizeof(__bf16);
    float*  attn_f     = (float*)((char*)d_ws + qkv_bytes);               // 1 MiB
    const size_t attn_bytes = (size_t)NBATCH * C_IN * C_IN * sizeof(float);
    __bf16* attn_b     = (__bf16*)((char*)attn_f + attn_bytes);           // 0.5 MiB

    (void)hipMemsetAsync(attn_f, 0, attn_bytes, stream);

    dim3 g1(NPIX / 64, OC / 128, NBATCH);          // 256 x 6 x 4
    qkv_gemm<<<g1, 256, 0, stream>>>(x, w, bias, qkv);

    dim3 g2(C_IN / 64, C_IN / 64, NBATCH * 8);     // 4 x 4 x 32 (8-way K split)
    qk_scores<<<g2, 256, 0, stream>>>(qkv, attn_f);

    softmax_rows<<<NBATCH * C_IN, 256, 0, stream>>>(attn_f, attn_b);

    dim3 g4(NPIX / 64, C_IN / 128, NBATCH);        // 256 x 2 x 4
    av_out<<<g4, 256, 0, stream>>>(attn_b, qkv, x, out);
}
